// GeoRwkvBlockV2_29583734735177
// MI455X (gfx1250) — compile-verified, matrix operand order checked by reading, not run on hardware
//
#include <hip/hip_runtime.h>
#include <math.h>

// GeoRwkvBlockV2 for MI455X (gfx1250, wave32, WMMA).
// All nine D^2-scale GEMMs run through v_wmma_f32_16x16x32_bf16 (bf16 in, f32 acc).
// WKV scan: one thread per (b,d) channel, 4096 channels, coalesced over d.

#define B_DIM 8
#define L_DIM 3969
#define D_DIM 512
#define SPLIT 102   // D/5
#define G0    104   // D - 4*SPLIT
#define RD2   (2 * D_DIM)   // router input length without gate_stats tail

typedef __attribute__((ext_vector_type(16))) __bf16 v16bf;
typedef __attribute__((ext_vector_type(8)))  float  v8f;

union FragBF { v16bf v; uint4 q[2]; };

__device__ __forceinline__ unsigned short f2bf(float f) {
  unsigned int u = __float_as_uint(f);
  u += 0x7fffu + ((u >> 16) & 1u);          // round-to-nearest-even
  return (unsigned short)(u >> 16);
}
__device__ __forceinline__ float sigf(float x)  { return 1.f / (1.f + expf(-x)); }
__device__ __forceinline__ float geluf(float x) { return 0.5f * x * (1.f + erff(x * 0.70710678118654752f)); }

template<int ACT>
__device__ __forceinline__ float act_apply(float x) {
  if (ACT == 1) return sigf(x);
  if (ACT == 2) return geluf(x);
  return x;
}

// ---------------------------------------------------------------------------
// WMMA GEMM: C[M,N] = act( A[M,K] @ W[N,K]^T  (+ A2 @ W2^T) (+ bias[N]) )
// A, W stored row-major with K contiguous -> identical fragment load pattern.
// One wave computes a 16x64 tile (4 accumulators), block = 4 waves = 128 thr.
// Fragment layout (16-bit A 16x32): lane l (0..15) holds row M=l%16,
// K base = (l/16)*8: q[0] = K[base..base+7], q[1] = K[16+base..16+base+7].
// ---------------------------------------------------------------------------
template<int ACT, bool HAS2, bool HASB, bool OUTBF>
__global__ __launch_bounds__(128)
void gemm_bf16_kernel(const unsigned short* __restrict__ A,
                      const unsigned short* __restrict__ W,
                      const unsigned short* __restrict__ A2,
                      const unsigned short* __restrict__ W2,
                      const float* __restrict__ bias,
                      float* __restrict__ Cf,
                      unsigned short* __restrict__ Cb,
                      int M, int N, int K)
{
  const int wave = threadIdx.x >> 5;
  const int lane = threadIdx.x & 31;
  const int lm   = lane & 15;
  const int half = lane >> 4;
  const int kb   = half * 8;

  const int m0 = blockIdx.x * 16;
  const int n0 = blockIdx.y * 256 + wave * 64;
  if (n0 >= N) return;                        // wave-uniform exit

  int arow = m0 + lm;
  if (arow >= M) arow = M - 1;                // clamp -> EXEC stays all-ones
  const size_t abase = (size_t)arow * K;

  size_t wbase[4];
#pragma unroll
  for (int j = 0; j < 4; ++j) wbase[j] = (size_t)(n0 + j * 16 + lm) * K;

  v8f c[4];
#pragma unroll
  for (int j = 0; j < 4; ++j) c[j] = (v8f){0.f,0.f,0.f,0.f,0.f,0.f,0.f,0.f};

  for (int k0 = 0; k0 < K; k0 += 32) {
    __builtin_prefetch(A + abase + k0 + 64, 0, 0);   // global_prefetch of A stream
    FragBF a;
    a.q[0] = *(const uint4*)(A + abase + k0 + kb);
    a.q[1] = *(const uint4*)(A + abase + k0 + kb + 16);
#pragma unroll
    for (int j = 0; j < 4; ++j) {
      FragBF b;
      b.q[0] = *(const uint4*)(W + wbase[j] + k0 + kb);
      b.q[1] = *(const uint4*)(W + wbase[j] + k0 + kb + 16);
      c[j] = __builtin_amdgcn_wmma_f32_16x16x32_bf16(
                 false, a.v, false, b.v, (short)0, c[j], false, false);
    }
    if (HAS2) {
      FragBF a2;
      a2.q[0] = *(const uint4*)(A2 + abase + k0 + kb);
      a2.q[1] = *(const uint4*)(A2 + abase + k0 + kb + 16);
#pragma unroll
      for (int j = 0; j < 4; ++j) {
        FragBF b2;
        b2.q[0] = *(const uint4*)(W2 + wbase[j] + k0 + kb);
        b2.q[1] = *(const uint4*)(W2 + wbase[j] + k0 + kb + 16);
        c[j] = __builtin_amdgcn_wmma_f32_16x16x32_bf16(
                   false, a2.v, false, b2.v, (short)0, c[j], false, false);
      }
    }
  }

  // C layout: VGPR i -> row m0 + half*8 + i, lane lm -> col offset
#pragma unroll
  for (int i = 0; i < 8; ++i) {
    int row = m0 + half * 8 + i;
    if (row < M) {
#pragma unroll
      for (int j = 0; j < 4; ++j) {
        int col = n0 + j * 16 + lm;
        float v = c[j][i];
        if (HASB) v += bias[col];
        v = act_apply<ACT>(v);
        if (OUTBF) Cb[(size_t)row * N + col] = f2bf(v);
        else       Cf[(size_t)row * N + col] = v;
      }
    }
  }
}

// ---------------------------------------------------------------------------
// fp32 -> bf16 weight conversion
// ---------------------------------------------------------------------------
__global__ void f32_to_bf16_kernel(const float* __restrict__ src,
                                   unsigned short* __restrict__ dst, int n) {
  int i = blockIdx.x * blockDim.x + threadIdx.x;
  if (i < n) dst[i] = f2bf(src[i]);
}

// ---------------------------------------------------------------------------
// block-wide sum over 256 threads
// ---------------------------------------------------------------------------
__device__ __forceinline__ float blk_reduce256(float v, float* red) {
  int tid = threadIdx.x;
  red[tid] = v;
  __syncthreads();
#pragma unroll
  for (int s = 128; s > 0; s >>= 1) {
    if (tid < s) red[tid] += red[tid + s];
    __syncthreads();
  }
  float r = red[0];
  __syncthreads();
  return r;
}

// ---------------------------------------------------------------------------
// spiral shift + LN(x), LN(l). One block per token.
// ---------------------------------------------------------------------------
__global__ __launch_bounds__(256)
void prep_ln_kernel(const float* __restrict__ x, const float* __restrict__ l,
                    const int* __restrict__ sidx, const unsigned char* __restrict__ svalid,
                    const float* __restrict__ gx, const float* __restrict__ bx,
                    const float* __restrict__ gl, const float* __restrict__ bl,
                    float* __restrict__ x_ln, float* __restrict__ l_ln,
                    unsigned short* __restrict__ x_ln_bf,
                    unsigned short* __restrict__ l_ln_bf)
{
  __shared__ float red[256];
  const int token = blockIdx.x;
  const int b = token / L_DIM;
  const int t = token - b * L_DIM;
  const size_t base = (size_t)token * D_DIM;
  const size_t bbase = (size_t)b * L_DIM * D_DIM;

  float vx[2], vl[2];
#pragma unroll
  for (int e = 0; e < 2; ++e) {
    int c = threadIdx.x + e * 256;
    float val;
    if (c < G0) {
      val = x[base + c];
    } else {
      int grp = (c - G0) / SPLIT;
      int src = sidx[grp * L_DIM + t];
      val = svalid[grp * L_DIM + t] ? x[bbase + (size_t)src * D_DIM + c] : 0.f;
    }
    vx[e] = val;
    vl[e] = l[base + c];
  }
  const float inv_d = 1.f / (float)D_DIM;
  float mux = blk_reduce256(vx[0] + vx[1], red) * inv_d;
  float vrx = blk_reduce256((vx[0]-mux)*(vx[0]-mux) + (vx[1]-mux)*(vx[1]-mux), red) * inv_d;
  float isx = rsqrtf(vrx + 1e-5f);
  float mul = blk_reduce256(vl[0] + vl[1], red) * inv_d;
  float vrl = blk_reduce256((vl[0]-mul)*(vl[0]-mul) + (vl[1]-mul)*(vl[1]-mul), red) * inv_d;
  float isl = rsqrtf(vrl + 1e-5f);
#pragma unroll
  for (int e = 0; e < 2; ++e) {
    int c = threadIdx.x + e * 256;
    float ox = (vx[e] - mux) * isx * gx[c] + bx[c];
    float ol = (vl[e] - mul) * isl * gl[c] + bl[c];
    x_ln[base + c] = ox;  x_ln_bf[base + c] = f2bf(ox);
    l_ln[base + c] = ol;  l_ln_bf[base + c] = f2bf(ol);
  }
}

// ---------------------------------------------------------------------------
// time-shift mix -> xk/xv/xr (bf16)
// ---------------------------------------------------------------------------
__global__ void tmix_kernel(const float* __restrict__ x_ln,
                            const float* __restrict__ tm_k,
                            const float* __restrict__ tm_v,
                            const float* __restrict__ tm_r,
                            unsigned short* __restrict__ xk,
                            unsigned short* __restrict__ xv,
                            unsigned short* __restrict__ xr)
{
  size_t i = (size_t)blockIdx.x * blockDim.x + threadIdx.x;
  const size_t ND = (size_t)B_DIM * L_DIM * D_DIM;
  if (i >= ND) return;
  int c = (int)(i & (D_DIM - 1));
  int token = (int)(i >> 9);
  int t = token % L_DIM;
  float cur  = x_ln[i];
  float prev = (t == 0) ? 0.f : x_ln[i - D_DIM];
  float mk = sigf(tm_k[c]), mv = sigf(tm_v[c]), mr = sigf(tm_r[c]);
  xk[i] = f2bf(cur * mk + prev * (1.f - mk));
  xv[i] = f2bf(cur * mv + prev * (1.f - mv));
  xr[i] = f2bf(cur * mr + prev * (1.f - mr));
}

// ---------------------------------------------------------------------------
// WKV scan: one thread per (b,d), sequential over L
// ---------------------------------------------------------------------------
__global__ void wkv_kernel(const float* __restrict__ k, const float* __restrict__ v,
                           const float* __restrict__ time_decay,
                           const float* __restrict__ time_first,
                           float* __restrict__ out)
{
  int idx = blockIdx.x * blockDim.x + threadIdx.x;
  if (idx >= B_DIM * D_DIM) return;
  int b = idx / D_DIM, d = idx - b * D_DIM;
  float w = -expf(time_decay[d]);
  float u = time_first[d];
  float a = 0.f, bb = 0.f, p = -1e38f;
  const size_t base = (size_t)b * L_DIM * D_DIM + d;
  for (int t = 0; t < L_DIM; ++t) {
    size_t o = base + (size_t)t * D_DIM;
    float kt = k[o], vt = v[o];
    float ww = u + kt;
    float q  = fmaxf(p, ww);
    float e1 = expf(p - q), e2 = expf(ww - q);
    out[o] = (e1 * a + e2 * vt) / (e1 * bb + e2);
    float ww2 = p + w;
    float q2  = fmaxf(ww2, kt);
    float e1b = expf(ww2 - q2), e2b = expf(kt - q2);
    a = e1b * a + e2b * vt;
    bb = e1b * bb + e2b;
    p = q2;
  }
}

// r * wkv -> bf16
__global__ void rwkv_mul_kernel(const float* __restrict__ r, const float* __restrict__ wkv,
                                unsigned short* __restrict__ out)
{
  size_t i = (size_t)blockIdx.x * blockDim.x + threadIdx.x;
  if (i >= (size_t)B_DIM * L_DIM * D_DIM) return;
  out[i] = f2bf(r[i] * wkv[i]);
}

// ---------------------------------------------------------------------------
// router-input means: one block per (b, j) output, j in [0, 2D)
// ---------------------------------------------------------------------------
__global__ __launch_bounds__(128)
void rin_kernel(const float* __restrict__ x_ln, const float* __restrict__ l_ln,
                float* __restrict__ rin)
{
  int b = blockIdx.x / RD2;
  int j = blockIdx.x - b * RD2;
  const float* src = (j < D_DIM)
      ? x_ln + (size_t)b * L_DIM * D_DIM + j
      : l_ln + (size_t)b * L_DIM * D_DIM + (j - D_DIM);
  float s = 0.f;
  for (int t = threadIdx.x; t < L_DIM; t += 128) s += src[(size_t)t * D_DIM];
  __shared__ float red[128];
  red[threadIdx.x] = s;
  __syncthreads();
#pragma unroll
  for (int st = 64; st > 0; st >>= 1) {
    if (threadIdx.x < st) red[threadIdx.x] += red[threadIdx.x + st];
    __syncthreads();
  }
  if (threadIdx.x == 0) rin[b * RD2 + j] = red[0] / (float)L_DIM;
}

// router MLP: 64 hidden units, one block per batch
__global__ __launch_bounds__(64)
void router_kernel(const float* __restrict__ rin, const float* __restrict__ gs,
                   const float* __restrict__ w1, const float* __restrict__ b1,
                   const float* __restrict__ w2, const float* __restrict__ b2,
                   float* __restrict__ alpha)
{
  int b = blockIdx.x, j = threadIdx.x;
  const int RDF = RD2 + 3;
  const float* rp = rin + b * RD2;
  const float* wp = w1 + j * RDF;
  float s = b1[j];
  for (int i = 0; i < RD2; ++i) s += rp[i] * wp[i];
  s += gs[b * 3 + 0] * wp[RD2 + 0];
  s += gs[b * 3 + 1] * wp[RD2 + 1];
  s += gs[b * 3 + 2] * wp[RD2 + 2];
  float hj = tanhf(s) * w2[j];
  __shared__ float red[64];
  red[j] = hj;
  __syncthreads();
#pragma unroll
  for (int st = 32; st > 0; st >>= 1) {
    if (j < st) red[j] += red[j + st];
    __syncthreads();
  }
  if (j == 0) alpha[b] = sigf(red[0] + b2[0]);
}

// ---------------------------------------------------------------------------
// h = x + beta*(alpha*att + (1-alpha)*fast); hc = LN(h). One block per token.
// ---------------------------------------------------------------------------
__global__ __launch_bounds__(256)
void combine_ln_kernel(const float* __restrict__ x, const float* __restrict__ att,
                       const float* __restrict__ fast, const float* __restrict__ alpha,
                       const float* __restrict__ gate_scale,
                       const float* __restrict__ gc, const float* __restrict__ bc,
                       float* __restrict__ h, float* __restrict__ hc)
{
  __shared__ float red[256];
  const int token = blockIdx.x;
  const int b = token / L_DIM;
  const float a = alpha[b];
  const float beta = log1pf(expf(gate_scale[0]));
  const size_t base = (size_t)token * D_DIM;
  float v[2];
#pragma unroll
  for (int e = 0; e < 2; ++e) {
    int c = threadIdx.x + e * 256;
    float hv = x[base + c] + beta * (a * att[base + c] + (1.f - a) * fast[base + c]);
    v[e] = hv;
    h[base + c] = hv;
  }
  const float inv_d = 1.f / (float)D_DIM;
  float mu = blk_reduce256(v[0] + v[1], red) * inv_d;
  float vr = blk_reduce256((v[0]-mu)*(v[0]-mu) + (v[1]-mu)*(v[1]-mu), red) * inv_d;
  float is = rsqrtf(vr + 1e-5f);
#pragma unroll
  for (int e = 0; e < 2; ++e) {
    int c = threadIdx.x + e * 256;
    hc[base + c] = (v[e] - mu) * is * gc[c] + bc[c];
  }
}

// channel-mix token shift -> xk2/xr2 (bf16)
__global__ void cmix_kernel(const float* __restrict__ hc,
                            const float* __restrict__ cm_mix_k,
                            const float* __restrict__ cm_mix_r,
                            unsigned short* __restrict__ xk2,
                            unsigned short* __restrict__ xr2)
{
  size_t i = (size_t)blockIdx.x * blockDim.x + threadIdx.x;
  if (i >= (size_t)B_DIM * L_DIM * D_DIM) return;
  int c = (int)(i & (D_DIM - 1));
  int token = (int)(i >> 9);
  int t = token % L_DIM;
  float cur  = hc[i];
  float prev = (t == 0) ? 0.f : hc[i - D_DIM];
  float ck = fminf(fmaxf(cm_mix_k[c], 0.f), 1.f);
  float cr = fminf(fmaxf(cm_mix_r[c], 0.f), 1.f);
  xk2[i] = f2bf(cur * ck + prev * (1.f - ck));
  xr2[i] = f2bf(cur * cr + prev * (1.f - cr));
}

// out = h + rr * vv
__global__ void final_kernel(const float* __restrict__ h, const float* __restrict__ rr,
                             const float* __restrict__ vv, float* __restrict__ out)
{
  size_t i = (size_t)blockIdx.x * blockDim.x + threadIdx.x;
  if (i >= (size_t)B_DIM * L_DIM * D_DIM) return;
  out[i] = h[i] + rr[i] * vv[i];
}

// ---------------------------------------------------------------------------
extern "C" void kernel_launch(void* const* d_in, const int* in_sizes, int n_in,
                              void* d_out, int out_size, void* d_ws, size_t ws_size,
                              hipStream_t stream)
{
  (void)in_sizes; (void)n_in; (void)out_size; (void)ws_size;
  const float* x          = (const float*)d_in[0];
  const float* l          = (const float*)d_in[1];
  const float* gate_stats = (const float*)d_in[2];
  const int*   shift_idx  = (const int*)d_in[3];
  const unsigned char* shift_valid = (const unsigned char*)d_in[4];  // jnp bool_
  const float* ln_x_g = (const float*)d_in[5];
  const float* ln_x_b = (const float*)d_in[6];
  const float* ln_l_g = (const float*)d_in[7];
  const float* ln_l_b = (const float*)d_in[8];
  const float* fm_w1  = (const float*)d_in[9];
  const float* fm_b1  = (const float*)d_in[10];
  const float* fm_w2  = (const float*)d_in[11];
  const float* fm_b2  = (const float*)d_in[12];
  const float* tm_k   = (const float*)d_in[13];
  const float* tm_v   = (const float*)d_in[14];
  const float* tm_r   = (const float*)d_in[15];
  const float* time_decay = (const float*)d_in[16];
  const float* time_first = (const float*)d_in[17];
  const float* w_rx   = (const float*)d_in[18];
  const float* w_kx   = (const float*)d_in[19];
  const float* w_vx   = (const float*)d_in[20];
  const float* w_rl   = (const float*)d_in[21];
  const float* w_kl   = (const float*)d_in[22];
  const float* w_out  = (const float*)d_in[23];
  const float* ln_c_g = (const float*)d_in[24];
  const float* ln_c_b = (const float*)d_in[25];
  const float* cm_mix_k = (const float*)d_in[26];
  const float* cm_mix_r = (const float*)d_in[27];
  const float* cm_key = (const float*)d_in[28];
  const float* cm_val = (const float*)d_in[29];
  const float* cm_rec = (const float*)d_in[30];
  const float* rt_w1  = (const float*)d_in[31];
  const float* rt_b1  = (const float*)d_in[32];
  const float* rt_w2  = (const float*)d_in[33];
  const float* rt_b2  = (const float*)d_in[34];
  const float* gate_scale = (const float*)d_in[35];
  float* out = (float*)d_out;

  const size_t MT = (size_t)B_DIM * L_DIM;       // 31752 tokens
  const size_t ND = MT * D_DIM;                  // 16 257 024 elements

  // ---- workspace carving (~0.95 GB total) ----
  char* ws = (char*)d_ws;
  size_t off = 0;
  auto alloc = [&](size_t bytes) -> void* {
    void* p = ws + off;
    off += (bytes + 255) & ~(size_t)255;
    return p;
  };
  float* x_ln    = (float*)alloc(ND * 4);
  float* l_ln    = (float*)alloc(ND * 4);
  float* kbuf    = (float*)alloc(ND * 4);
  float* vbuf    = (float*)alloc(ND * 4);
  float* wkvbuf  = (float*)alloc(ND * 4);
  float* rbuf    = (float*)alloc(ND * 4);
  float* attbuf  = (float*)alloc(ND * 4);
  float* fastbuf = (float*)alloc(ND * 4);
  float* hbuf    = (float*)alloc(ND * 4);
  float* hcbuf   = (float*)alloc(ND * 4);
  unsigned short* x_ln_bf = (unsigned short*)alloc(ND * 2);
  unsigned short* l_ln_bf = (unsigned short*)alloc(ND * 2);
  unsigned short* xk_bf   = (unsigned short*)alloc(ND * 2);
  unsigned short* xv_bf   = (unsigned short*)alloc(ND * 2);
  unsigned short* xr_bf   = (unsigned short*)alloc(ND * 2);
  unsigned short* rwkv_bf = (unsigned short*)alloc(ND * 2);
  unsigned short* big_bf  = (unsigned short*)alloc(MT * 2048 * 2);  // fast1 (M x 1536) then cm1 (M x 2048)
  // bf16 weights
  unsigned short* wrx_bf  = (unsigned short*)alloc((size_t)D_DIM * D_DIM * 2);
  unsigned short* wkx_bf  = (unsigned short*)alloc((size_t)D_DIM * D_DIM * 2);
  unsigned short* wvx_bf  = (unsigned short*)alloc((size_t)D_DIM * D_DIM * 2);
  unsigned short* wrl_bf  = (unsigned short*)alloc((size_t)D_DIM * D_DIM * 2);
  unsigned short* wkl_bf  = (unsigned short*)alloc((size_t)D_DIM * D_DIM * 2);
  unsigned short* wout_bf = (unsigned short*)alloc((size_t)D_DIM * D_DIM * 2);
  unsigned short* crec_bf = (unsigned short*)alloc((size_t)D_DIM * D_DIM * 2);
  unsigned short* fw1_bf  = (unsigned short*)alloc((size_t)3 * D_DIM * D_DIM * 2);
  unsigned short* fw2_bf  = (unsigned short*)alloc((size_t)3 * D_DIM * D_DIM * 2);
  unsigned short* ckey_bf = (unsigned short*)alloc((size_t)4 * D_DIM * D_DIM * 2);
  unsigned short* cval_bf = (unsigned short*)alloc((size_t)4 * D_DIM * D_DIM * 2);
  float* rin    = (float*)alloc((size_t)B_DIM * RD2 * 4);
  float* alphab = (float*)alloc((size_t)B_DIM * 4);
  // buffer reuse after the attention/fast paths are consumed:
  unsigned short* xk2_bf = xk_bf;
  unsigned short* xr2_bf = xv_bf;
  float* vvbuf = kbuf;
  float* rrbuf = vbuf;

  // ---- 1. weight conversion fp32 -> bf16 ----
  const int DD  = D_DIM * D_DIM;
  const int DD3 = 3 * DD;
  const int DD4 = 4 * DD;
  auto conv = [&](const float* s, unsigned short* d, int n) {
    f32_to_bf16_kernel<<<(n + 255) / 256, 256, 0, stream>>>(s, d, n);
  };
  conv(w_rx, wrx_bf, DD);   conv(w_kx, wkx_bf, DD);   conv(w_vx, wvx_bf, DD);
  conv(w_rl, wrl_bf, DD);   conv(w_kl, wkl_bf, DD);   conv(w_out, wout_bf, DD);
  conv(cm_rec, crec_bf, DD);
  conv(fm_w1, fw1_bf, DD3); conv(fm_w2, fw2_bf, DD3);
  conv(cm_key, ckey_bf, DD4); conv(cm_val, cval_bf, DD4);

  // ---- 2. spiral shift + layernorms ----
  prep_ln_kernel<<<(unsigned)MT, 256, 0, stream>>>(
      x, l, shift_idx, shift_valid, ln_x_g, ln_x_b, ln_l_g, ln_l_b,
      x_ln, l_ln, x_ln_bf, l_ln_bf);

  // ---- 3. token-shift mixes ----
  const unsigned EW = (unsigned)((ND + 255) / 256);
  tmix_kernel<<<EW, 256, 0, stream>>>(x_ln, tm_k, tm_v, tm_r, xk_bf, xv_bf, xr_bf);

  // ---- 4. attention-path GEMMs ----
  const unsigned GM = (unsigned)((MT + 15) / 16);
  gemm_bf16_kernel<0, true,  false, false><<<dim3(GM, 2), 128, 0, stream>>>(
      xk_bf, wkx_bf, l_ln_bf, wkl_bf, nullptr, kbuf, nullptr, (int)MT, D_DIM, D_DIM);
  gemm_bf16_kernel<0, false, false, false><<<dim3(GM, 2), 128, 0, stream>>>(
      xv_bf, wvx_bf, nullptr, nullptr, nullptr, vbuf, nullptr, (int)MT, D_DIM, D_DIM);
  gemm_bf16_kernel<1, true,  false, false><<<dim3(GM, 2), 128, 0, stream>>>(
      xr_bf, wrx_bf, l_ln_bf, wrl_bf, nullptr, rbuf, nullptr, (int)MT, D_DIM, D_DIM);

  // ---- 5. WKV scan ----
  wkv_kernel<<<(B_DIM * D_DIM + 255) / 256, 256, 0, stream>>>(
      kbuf, vbuf, time_decay, time_first, wkvbuf);
  rwkv_mul_kernel<<<EW, 256, 0, stream>>>(rbuf, wkvbuf, rwkv_bf);
  gemm_bf16_kernel<0, false, false, false><<<dim3(GM, 2), 128, 0, stream>>>(
      rwkv_bf, wout_bf, nullptr, nullptr, nullptr, attbuf, nullptr, (int)MT, D_DIM, D_DIM);

  // ---- 6. fast path: gelu(x_ln @ fm_w1^T + b1) @ fm_w2^T + b2 ----
  gemm_bf16_kernel<2, false, true, true><<<dim3(GM, 6), 128, 0, stream>>>(
      x_ln_bf, fw1_bf, nullptr, nullptr, fm_b1, nullptr, big_bf, (int)MT, 3 * D_DIM, D_DIM);
  gemm_bf16_kernel<0, false, true, false><<<dim3(GM, 2), 128, 0, stream>>>(
      big_bf, fw2_bf, nullptr, nullptr, fm_b2, fastbuf, nullptr, (int)MT, D_DIM, 3 * D_DIM);

  // ---- 7. router ----
  rin_kernel<<<B_DIM * RD2, 128, 0, stream>>>(x_ln, l_ln, rin);
  router_kernel<<<B_DIM, 64, 0, stream>>>(rin, gate_stats, rt_w1, rt_b1, rt_w2, rt_b2, alphab);

  // ---- 8. combine + LN ----
  combine_ln_kernel<<<(unsigned)MT, 256, 0, stream>>>(
      x, attbuf, fastbuf, alphab, gate_scale, ln_c_g, ln_c_b, hbuf, hcbuf);

  // ---- 9. channel mix ----
  cmix_kernel<<<EW, 256, 0, stream>>>(hcbuf, cm_mix_k, cm_mix_r, xk2_bf, xr2_bf);
  gemm_bf16_kernel<2, false, false, true><<<dim3(GM, 8), 128, 0, stream>>>(
      xk2_bf, ckey_bf, nullptr, nullptr, nullptr, nullptr, big_bf, (int)MT, 4 * D_DIM, D_DIM);
  gemm_bf16_kernel<0, false, false, false><<<dim3(GM, 2), 128, 0, stream>>>(
      big_bf, cval_bf, nullptr, nullptr, nullptr, vvbuf, nullptr, (int)MT, D_DIM, 4 * D_DIM);
  gemm_bf16_kernel<1, false, false, false><<<dim3(GM, 2), 128, 0, stream>>>(
      xr2_bf, crec_bf, nullptr, nullptr, nullptr, rrbuf, nullptr, (int)MT, D_DIM, D_DIM);

  // ---- 10. output ----
  final_kernel<<<EW, 256, 0, stream>>>(hbuf, rrbuf, vvbuf, out);
}
